// GCN_37383395344933
// MI455X (gfx1250) — matrix operand order repaired and checked by reading, not compiled, not run on hardware
//
#include <hip/hip_runtime.h>
#include <hip/hip_bf16.h>

// ---------------------------------------------------------------------------
// CDNA5 (gfx1250) implementation. wave32. Matrix math via
// v_wmma_f32_16x16x32_f16 (f16 inputs, f32 accumulate).
//
// Fragment layouts (per CDNA5 ISA 7.12.2):
//  A (16x32 f16):  lane = m + 16*k[3];  lane's 16 halves:
//                  e0..7  -> k0 + 8*lh + (0..7)
//                  e8..15 -> k0 + 16 + 8*lh + (0..7)
//  B (32x16 f16):  lane = n + 16*hi;  lane's 16 halves:
//                  e0..7 -> k0 + 8*hi + (0..7), e8..15 -> k0 + 16 + 8*hi + (0..7)
//  C/D (16x16 f32): vgpr j holds (m = j + 8*(lane>>4), n = lane&15)
// ---------------------------------------------------------------------------

typedef __attribute__((ext_vector_type(16))) _Float16 v16h;
typedef __attribute__((ext_vector_type(8)))  _Float16 v8h;
typedef __attribute__((ext_vector_type(4)))  _Float16 v4h;
typedef __attribute__((ext_vector_type(8)))  float    v8f;

#define DEV static __device__ __forceinline__

// Branchless exact-gelu: 0.5*x*(1+erf(x/sqrt2)) with Abramowitz-Stegun 7.1.26
// erf approximation (|err|<=1.5e-7), using raw v_rcp_f32 / v_exp_f32.
DEV float geluf(float x) {
  float z = x * 0.70710678118654752440f;
  float az = fabsf(z);
  float t = __builtin_amdgcn_rcpf(1.0f + 0.3275911f * az);
  float p = t * (0.254829592f +
            t * (-0.284496736f +
            t * (1.421413741f +
            t * (-1.453152027f +
            t * 1.061405429f))));
  float e = __builtin_amdgcn_exp2f(-az * az * 1.4426950408889634f);
  float er = 1.0f - p * e;
  unsigned bits = (__float_as_uint(er) & 0x7fffffffu) |
                  (__float_as_uint(z) & 0x80000000u);
  return 0.5f * x * (1.0f + __uint_as_float(bits));
}

DEV v8f vzero() {
  v8f z;
#pragma unroll
  for (int i = 0; i < 8; ++i) z[i] = 0.0f;
  return z;
}

DEV v8f wmma16(v16h a, v16h b, v8f c) {
  return __builtin_amdgcn_wmma_f32_16x16x32_f16(false, a, false, b,
                                                (short)0, c, false, false);
}

// A fragment from an fp32 global row (convert to f16 on the fly)
DEV v16h afrag_g32(const float* __restrict__ row, int k0, int lh) {
  const float4* p0 = reinterpret_cast<const float4*>(row + k0 + 8 * lh);
  const float4* p1 = reinterpret_cast<const float4*>(row + k0 + 16 + 8 * lh);
  float4 a = p0[0], b = p0[1], c = p1[0], d = p1[1];
  v16h r;
  r[0]  = (_Float16)a.x; r[1]  = (_Float16)a.y; r[2]  = (_Float16)a.z; r[3]  = (_Float16)a.w;
  r[4]  = (_Float16)b.x; r[5]  = (_Float16)b.y; r[6]  = (_Float16)b.z; r[7]  = (_Float16)b.w;
  r[8]  = (_Float16)c.x; r[9]  = (_Float16)c.y; r[10] = (_Float16)c.z; r[11] = (_Float16)c.w;
  r[12] = (_Float16)d.x; r[13] = (_Float16)d.y; r[14] = (_Float16)d.z; r[15] = (_Float16)d.w;
  return r;
}

// same, with per-row scale (folds 1/deg into the mean aggregation)
DEV v16h afrag_g32s(const float* __restrict__ row, int k0, int lh, float s) {
  const float4* p0 = reinterpret_cast<const float4*>(row + k0 + 8 * lh);
  const float4* p1 = reinterpret_cast<const float4*>(row + k0 + 16 + 8 * lh);
  float4 a = p0[0], b = p0[1], c = p1[0], d = p1[1];
  v16h r;
  r[0]  = (_Float16)(a.x * s); r[1]  = (_Float16)(a.y * s); r[2]  = (_Float16)(a.z * s); r[3]  = (_Float16)(a.w * s);
  r[4]  = (_Float16)(b.x * s); r[5]  = (_Float16)(b.y * s); r[6]  = (_Float16)(b.z * s); r[7]  = (_Float16)(b.w * s);
  r[8]  = (_Float16)(c.x * s); r[9]  = (_Float16)(c.y * s); r[10] = (_Float16)(c.z * s); r[11] = (_Float16)(c.w * s);
  r[12] = (_Float16)(d.x * s); r[13] = (_Float16)(d.y * s); r[14] = (_Float16)(d.z * s); r[15] = (_Float16)(d.w * s);
  return r;
}

// A fragment from an f16 LDS row
DEV v16h afrag_l16(const _Float16* row, int k0, int lh) {
  v8h a = *reinterpret_cast<const v8h*>(row + k0 + 8 * lh);
  v8h b = *reinterpret_cast<const v8h*>(row + k0 + 16 + 8 * lh);
  v16h r;
#pragma unroll
  for (int i = 0; i < 8; ++i) { r[i] = a[i]; r[8 + i] = b[i]; }
  return r;
}

// B fragment from pre-converted f16 weights W[dout][din]; B[k][n] = W[n][k]
DEV v16h bfrag16(const _Float16* __restrict__ w, int din, int n, int k0, int hi) {
  const _Float16* p = w + (size_t)n * din + k0 + 8 * hi;
  v8h a = *reinterpret_cast<const v8h*>(p);
  v8h b = *reinterpret_cast<const v8h*>(p + 16);
  v16h r;
#pragma unroll
  for (int i = 0; i < 8; ++i) { r[i] = a[i]; r[8 + i] = b[i]; }
  return r;
}

// float4 -> packed 4x f16 store (8B)
DEV void st4h(_Float16* p, float4 v) {
  v4h h;
  h[0] = (_Float16)v.x; h[1] = (_Float16)v.y;
  h[2] = (_Float16)v.z; h[3] = (_Float16)v.w;
  *reinterpret_cast<v4h*>(p) = h;
}

// ---------------------------------------------------------------------------
// small utility kernels
// ---------------------------------------------------------------------------
__global__ void k_cvt(const float* __restrict__ s, _Float16* __restrict__ d, int n) {
  int t = blockIdx.x * 256 + threadIdx.x;
  if (t < n) d[t] = (_Float16)s[t];
}

__global__ void k_zero(float* __restrict__ p, int n) {
  int t = blockIdx.x * 256 + threadIdx.x;
  if (t < n) p[t] = 0.0f;
}

__global__ void k_deg(const int* __restrict__ dst, float* __restrict__ deg, int E) {
  int t = blockIdx.x * 256 + threadIdx.x;
  if (t < E) atomicAdd(&deg[dst[t]], 1.0f);
}

// segment_sum of h[src] into agg[dst]; 8 threads per edge, 8 channels each
__global__ void k_scatter(const int* __restrict__ src, const int* __restrict__ dst,
                          const float* __restrict__ h, float* __restrict__ agg, int E) {
  int t = blockIdx.x * 256 + threadIdx.x;
  int e = t >> 3;
  if (e < E) {
    int s = src[e], d = dst[e];
    int c0 = (t & 7) * 8;
    const float* hp = h + (size_t)s * 64 + c0;
    float* ap = agg + (size_t)d * 64 + c0;
#pragma unroll
    for (int i = 0; i < 8; ++i) atomicAdd(ap + i, hp[i]);
  }
}

// order-preserving float->uint for atomic max; init = encode(-inf)
__global__ void k_pool_init(unsigned* __restrict__ pu) {
  pu[blockIdx.x * 256 + threadIdx.x] = 0x007FFFFFu;
}
__global__ void k_pool(const float* __restrict__ h, const int* __restrict__ batch,
                       unsigned* __restrict__ pu, int n) {
  int t = blockIdx.x * 256 + threadIdx.x;
  if (t < n * 64) {
    int node = t >> 6, c = t & 63;
    unsigned b = __float_as_uint(h[t]);
    unsigned u = (b & 0x80000000u) ? ~b : (b | 0x80000000u);
    atomicMax(&pu[(size_t)batch[node] * 64 + c], u);
  }
}
__global__ void k_pool_dec(const unsigned* __restrict__ pu, float* __restrict__ g) {
  int t = blockIdx.x * 256 + threadIdx.x;
  unsigned u = pu[t];
  unsigned b = (u & 0x80000000u) ? (u & 0x7FFFFFFFu) : ~u;
  g[t] = __uint_as_float(b);
}

// ---------------------------------------------------------------------------
// Fused embedder: per block = 16 nodes (64 rows of x).  h0[n,:] =
//   mlp( sum_s gelu(mlp(x[n,s,:], emb_mlp)), emb_rho )
// ---------------------------------------------------------------------------
__global__ __launch_bounds__(128) void k_embed(
    const float* __restrict__ x,
    const _Float16* __restrict__ Wein, const _Float16* __restrict__ Wepr,
    const _Float16* __restrict__ Wehd, const _Float16* __restrict__ Weot,
    const float* __restrict__ bein, const float* __restrict__ bepr,
    const float* __restrict__ behd, const float* __restrict__ beot,
    const _Float16* __restrict__ Wrin, const _Float16* __restrict__ Wrpr,
    const _Float16* __restrict__ Wrhd, const _Float16* __restrict__ Wrot,
    const float* __restrict__ brin, const float* __restrict__ brpr,
    const float* __restrict__ brhd, const float* __restrict__ brot,
    float* __restrict__ h0) {
  __shared__ _Float16 qh[16 * 512];      // pooled node features (f16), 16KB
  __shared__ _Float16 stg[4][16 * 64];   // per-wave staging, 8KB
  __shared__ _Float16 sr[16 * 64];       // rho staging, 2KB

  const int lane = threadIdx.x & 31, wv = threadIdx.x >> 5;
  const int m = lane & 15, lh = lane >> 4;
  const int cn = m, ch = lh;
  const int node0 = blockIdx.x * 16;
  const int wrow = node0 * 4 + wv * 16;

  // ---- phase 1: a = gelu(x@Win^T + b); skip = x@Wproj^T (K=512, N=64) ----
  v8f aA[4], aS[4];
#pragma unroll
  for (int nt = 0; nt < 4; ++nt) { aA[nt] = vzero(); aS[nt] = vzero(); }
  const float* xrow = x + (size_t)(wrow + m) * 512;
  for (int kb = 0; kb < 16; ++kb) {
    v16h af = afrag_g32(xrow, kb * 32, lh);
#pragma unroll
    for (int nt = 0; nt < 4; ++nt) {
      aA[nt] = wmma16(af, bfrag16(Wein, 512, nt * 16 + cn, kb * 32, lh), aA[nt]);
      aS[nt] = wmma16(af, bfrag16(Wepr, 512, nt * 16 + cn, kb * 32, lh), aS[nt]);
    }
  }
#pragma unroll
  for (int nt = 0; nt < 4; ++nt) {
    float bi = bein[nt * 16 + cn];
#pragma unroll
    for (int j = 0; j < 8; ++j)
      stg[wv][(j + 8 * ch) * 64 + nt * 16 + cn] = (_Float16)geluf(aA[nt][j] + bi);
  }

  // ---- hid GEMM (K=64) ----
  v8f aH[4];
#pragma unroll
  for (int nt = 0; nt < 4; ++nt) aH[nt] = vzero();
  const _Float16* srow = &stg[wv][m * 64];
#pragma unroll
  for (int kb = 0; kb < 2; ++kb) {
    v16h af = afrag_l16(srow, kb * 32, lh);
#pragma unroll
    for (int nt = 0; nt < 4; ++nt)
      aH[nt] = wmma16(af, bfrag16(Wehd, 64, nt * 16 + cn, kb * 32, lh), aH[nt]);
  }
  // t3 = (skip + bproj) + gelu(hid + bhid) -> stg (same-wave DS ordering)
#pragma unroll
  for (int nt = 0; nt < 4; ++nt) {
    float bp = bepr[nt * 16 + cn], bh = behd[nt * 16 + cn];
#pragma unroll
    for (int j = 0; j < 8; ++j)
      stg[wv][(j + 8 * ch) * 64 + nt * 16 + cn] =
          (_Float16)((aS[nt][j] + bp) + geluf(aH[nt][j] + bh));
  }

  // ---- lin_out (K=64, N=512) + gelu + sum over S (in-register) ----
  v16h t3f0 = afrag_l16(srow, 0, lh);
  v16h t3f1 = afrag_l16(srow, 32, lh);
  const int gn0 = wv * 4 + ch * 2;
  for (int ng = 0; ng < 8; ++ng) {
    v8f acc[4];
#pragma unroll
    for (int nt = 0; nt < 4; ++nt) acc[nt] = vzero();
#pragma unroll
    for (int nt = 0; nt < 4; ++nt) {
      acc[nt] = wmma16(t3f0, bfrag16(Weot, 64, ng * 64 + nt * 16 + cn, 0, lh), acc[nt]);
      acc[nt] = wmma16(t3f1, bfrag16(Weot, 64, ng * 64 + nt * 16 + cn, 32, lh), acc[nt]);
    }
#pragma unroll
    for (int nt = 0; nt < 4; ++nt) {
      float bo = beot[ng * 64 + nt * 16 + cn];
      float s0 = geluf(acc[nt][0] + bo) + geluf(acc[nt][1] + bo) +
                 geluf(acc[nt][2] + bo) + geluf(acc[nt][3] + bo);
      float s1 = geluf(acc[nt][4] + bo) + geluf(acc[nt][5] + bo) +
                 geluf(acc[nt][6] + bo) + geluf(acc[nt][7] + bo);
      qh[gn0 * 512 + ng * 64 + nt * 16 + cn] = (_Float16)s0;
      qh[(gn0 + 1) * 512 + ng * 64 + nt * 16 + cn] = (_Float16)s1;
    }
  }
  __syncthreads();

  // ---- rho MLP on 16 pooled nodes; wave wv owns output n-tile wv ----
  v8f rA = vzero(), rS = vzero();
  const _Float16* qrow = &qh[m * 512];
  for (int kb = 0; kb < 16; ++kb) {
    v16h af = afrag_l16(qrow, kb * 32, lh);
    rA = wmma16(af, bfrag16(Wrin, 512, wv * 16 + cn, kb * 32, lh), rA);
    rS = wmma16(af, bfrag16(Wrpr, 512, wv * 16 + cn, kb * 32, lh), rS);
  }
  {
    float bi = brin[wv * 16 + cn];
#pragma unroll
    for (int j = 0; j < 8; ++j)
      sr[(j + 8 * ch) * 64 + wv * 16 + cn] = (_Float16)geluf(rA[j] + bi);
  }
  __syncthreads();
  v8f rH = vzero();
  const _Float16* srrow = &sr[m * 64];
#pragma unroll
  for (int kb = 0; kb < 2; ++kb) {
    v16h af = afrag_l16(srrow, kb * 32, lh);
    rH = wmma16(af, bfrag16(Wrhd, 64, wv * 16 + cn, kb * 32, lh), rH);
  }
  __syncthreads();  // all waves done reading sr before overwrite
  {
    float bp = brpr[wv * 16 + cn], bh = brhd[wv * 16 + cn];
#pragma unroll
    for (int j = 0; j < 8; ++j)
      sr[(j + 8 * ch) * 64 + wv * 16 + cn] =
          (_Float16)((rS[j] + bp) + geluf(rH[j] + bh));
  }
  __syncthreads();
  v8f rO = vzero();
#pragma unroll
  for (int kb = 0; kb < 2; ++kb) {
    v16h af = afrag_l16(srrow, kb * 32, lh);
    rO = wmma16(af, bfrag16(Wrot, 64, wv * 16 + cn, kb * 32, lh), rO);
  }
  {
    float bo = brot[wv * 16 + cn];
#pragma unroll
    for (int j = 0; j < 8; ++j)
      h0[(size_t)(node0 + j + 8 * ch) * 64 + wv * 16 + cn] = rO[j] + bo;
  }
}

// ---------------------------------------------------------------------------
// SAGE conv: h_out = h + gelu( (agg/deg)@Wl^T + bl + h@Wr^T )
// ---------------------------------------------------------------------------
__global__ __launch_bounds__(128) void k_conv(
    const float* __restrict__ hin, const float* __restrict__ agg,
    const float* __restrict__ deg,
    const _Float16* __restrict__ Wl, const float* __restrict__ bl,
    const _Float16* __restrict__ Wr,
    float* __restrict__ hout, int n) {
  const int lane = threadIdx.x & 31, wv = threadIdx.x >> 5;
  const int m = lane & 15, lh = lane >> 4;
  const int cn = m, ch = lh;
  const int row0 = blockIdx.x * 64 + wv * 16;
  int rowm = row0 + m;
  if (rowm > n - 1) rowm = n - 1;
  float rd = __builtin_amdgcn_rcpf(fmaxf(deg[rowm], 1.0f));
  const float* arow = agg + (size_t)rowm * 64;
  const float* hrow = hin + (size_t)rowm * 64;

  v8f acc[4];
#pragma unroll
  for (int nt = 0; nt < 4; ++nt) acc[nt] = vzero();
#pragma unroll
  for (int kb = 0; kb < 2; ++kb) {
    v16h aa = afrag_g32s(arow, kb * 32, lh, rd);
    v16h ah = afrag_g32(hrow, kb * 32, lh);
#pragma unroll
    for (int nt = 0; nt < 4; ++nt) {
      acc[nt] = wmma16(aa, bfrag16(Wl, 64, nt * 16 + cn, kb * 32, lh), acc[nt]);
      acc[nt] = wmma16(ah, bfrag16(Wr, 64, nt * 16 + cn, kb * 32, lh), acc[nt]);
    }
  }
#pragma unroll
  for (int nt = 0; nt < 4; ++nt) {
    float b = bl[nt * 16 + cn];
#pragma unroll
    for (int j = 0; j < 8; ++j) {
      int rr = row0 + j + 8 * ch;
      if (rr < n) {
        size_t o = (size_t)rr * 64 + nt * 16 + cn;
        hout[o] = hin[o] + geluf(acc[nt][j] + b);
      }
    }
  }
}

// ---------------------------------------------------------------------------
// Head: out[q] = head(xy) - head(yx), 64 queries/block, 4 waves x 16 queries
// ---------------------------------------------------------------------------
__global__ __launch_bounds__(128) void k_head(
    const float* __restrict__ h, const float* __restrict__ gemb,
    const int* __restrict__ srci, const int* __restrict__ dsti,
    const int* __restrict__ batch,
    const _Float16* __restrict__ W1in, const float* __restrict__ b1in,
    const _Float16* __restrict__ W1pr, const float* __restrict__ b1pr,
    const _Float16* __restrict__ W1hd, const float* __restrict__ b1hd,
    const _Float16* __restrict__ W1ot, const float* __restrict__ b1ot,
    const _Float16* __restrict__ W2in, const float* __restrict__ b2in,
    const _Float16* __restrict__ W2hd, const float* __restrict__ b2hd,
    const float* __restrict__ W2ot,
    float* __restrict__ out) {
  __shared__ _Float16 q16[64 * 192];        // 24 KB input tile (f16)
  __shared__ _Float16 stg[4][2][16 * 128];  // 32 KB per-wave staging

  const int q0 = blockIdx.x * 64;
  {  // build q tile: [h[src] | h[dst] | g] as f16 (vectorized b128 loads)
    int r = threadIdx.x >> 1, part = threadIdx.x & 1;
    int qi = q0 + r;
    int s = srci[qi], d = dsti[qi];
    const float4* hs = reinterpret_cast<const float4*>(h + (size_t)s * 64);
    const float4* hd = reinterpret_cast<const float4*>(h + (size_t)d * 64);
    const float4* gg = reinterpret_cast<const float4*>(gemb + (size_t)batch[s] * 64);
    _Float16* qr = &q16[r * 192];
    if (part == 0) {
#pragma unroll
      for (int i = 0; i < 16; ++i) st4h(qr + 4 * i, hs[i]);
#pragma unroll
      for (int i = 0; i < 8; ++i) st4h(qr + 64 + 4 * i, hd[i]);
    } else {
#pragma unroll
      for (int i = 0; i < 8; ++i) st4h(qr + 96 + 4 * i, hd[8 + i]);
#pragma unroll
      for (int i = 0; i < 16; ++i) st4h(qr + 128 + 4 * i, gg[i]);
    }
  }
  __syncthreads();

  const int lane = threadIdx.x & 31, wv = threadIdx.x >> 5;
  const int m = lane & 15, lh = lane >> 4;
  const int cn = m, ch = lh;
  const _Float16* qrow = &q16[(wv * 16 + m) * 192];
  _Float16* s0 = &stg[wv][0][0];
  _Float16* s1 = &stg[wv][1][0];
  const _Float16* s0row = s0 + m * 128;
  const _Float16* s1row = s1 + m * 128;

  float res[8];
#pragma unroll
  for (int j = 0; j < 8; ++j) res[j] = 0.0f;

  for (int var = 0; var < 2; ++var) {
    // lin1: a = gelu(q@W1in^T+b); skip = q@W1pr^T (K=192, N=128)
    v8f aA[8], aS[8];
#pragma unroll
    for (int nt = 0; nt < 8; ++nt) { aA[nt] = vzero(); aS[nt] = vzero(); }
#pragma unroll
    for (int ks = 0; ks < 6; ++ks) {
      int half = ks >> 1;                                       // 0,0,1,1,2,2 (xy)
      if (var) half = (half == 0) ? 1 : ((half == 1) ? 0 : 2);  // yx swaps src/dst
      v16h af = afrag_l16(qrow, half * 64 + (ks & 1) * 32, lh);
#pragma unroll
      for (int nt = 0; nt < 8; ++nt) {
        aA[nt] = wmma16(af, bfrag16(W1in, 192, nt * 16 + cn, ks * 32, lh), aA[nt]);
        aS[nt] = wmma16(af, bfrag16(W1pr, 192, nt * 16 + cn, ks * 32, lh), aS[nt]);
      }
    }
#pragma unroll
    for (int nt = 0; nt < 8; ++nt) {
      float bi = b1in[nt * 16 + cn];
#pragma unroll
      for (int j = 0; j < 8; ++j)
        s0[(j + 8 * ch) * 128 + nt * 16 + cn] = (_Float16)geluf(aA[nt][j] + bi);
    }
    // hid (K=128)
    v8f aH[8];
#pragma unroll
    for (int nt = 0; nt < 8; ++nt) aH[nt] = vzero();
#pragma unroll
    for (int kb = 0; kb < 4; ++kb) {
      v16h af = afrag_l16(s0row, kb * 32, lh);
#pragma unroll
      for (int nt = 0; nt < 8; ++nt)
        aH[nt] = wmma16(af, bfrag16(W1hd, 128, nt * 16 + cn, kb * 32, lh), aH[nt]);
    }
    // t3 = skip + gelu(hid) -> s0
#pragma unroll
    for (int nt = 0; nt < 8; ++nt) {
      float bp = b1pr[nt * 16 + cn], bh = b1hd[nt * 16 + cn];
#pragma unroll
      for (int j = 0; j < 8; ++j)
        s0[(j + 8 * ch) * 128 + nt * 16 + cn] =
            (_Float16)((aS[nt][j] + bp) + geluf(aH[nt][j] + bh));
    }
    // r = relu(t3@W1ot^T + b) -> s0
    v8f aR[8];
#pragma unroll
    for (int nt = 0; nt < 8; ++nt) aR[nt] = vzero();
#pragma unroll
    for (int kb = 0; kb < 4; ++kb) {
      v16h af = afrag_l16(s0row, kb * 32, lh);
#pragma unroll
      for (int nt = 0; nt < 8; ++nt)
        aR[nt] = wmma16(af, bfrag16(W1ot, 128, nt * 16 + cn, kb * 32, lh), aR[nt]);
    }
#pragma unroll
    for (int nt = 0; nt < 8; ++nt) {
      float bo = b1ot[nt * 16 + cn];
#pragma unroll
      for (int j = 0; j < 8; ++j)
        s0[(j + 8 * ch) * 128 + nt * 16 + cn] =
            (_Float16)fmaxf(aR[nt][j] + bo, 0.0f);
    }
    // lin2: a2 = gelu(r@W2in^T+b) -> s1
    v8f c2[8];
#pragma unroll
    for (int nt = 0; nt < 8; ++nt) c2[nt] = vzero();
#pragma unroll
    for (int kb = 0; kb < 4; ++kb) {
      v16h af = afrag_l16(s0row, kb * 32, lh);
#pragma unroll
      for (int nt = 0; nt < 8; ++nt)
        c2[nt] = wmma16(af, bfrag16(W2in, 128, nt * 16 + cn, kb * 32, lh), c2[nt]);
    }
#pragma unroll
    for (int nt = 0; nt < 8; ++nt) {
      float bi = b2in[nt * 16 + cn];
#pragma unroll
      for (int j = 0; j < 8; ++j)
        s1[(j + 8 * ch) * 128 + nt * 16 + cn] = (_Float16)geluf(c2[nt][j] + bi);
    }
    v8f c3[8];
#pragma unroll
    for (int nt = 0; nt < 8; ++nt) c3[nt] = vzero();
#pragma unroll
    for (int kb = 0; kb < 4; ++kb) {
      v16h af = afrag_l16(s1row, kb * 32, lh);
#pragma unroll
      for (int nt = 0; nt < 8; ++nt)
        c3[nt] = wmma16(af, bfrag16(W2hd, 128, nt * 16 + cn, kb * 32, lh), c3[nt]);
    }
    // final 128->1: u[m] = sum_n (r + gelu(c3+bh2))[m][n] * W2ot[n]
    float part[8];
#pragma unroll
    for (int j = 0; j < 8; ++j) part[j] = 0.0f;
#pragma unroll
    for (int nt = 0; nt < 8; ++nt) {
      float bh = b2hd[nt * 16 + cn];
      float w2 = W2ot[nt * 16 + cn];
#pragma unroll
      for (int j = 0; j < 8; ++j) {
        float rv = (float)s0[(j + 8 * ch) * 128 + nt * 16 + cn];
        part[j] += (rv + geluf(c3[nt][j] + bh)) * w2;
      }
    }
#pragma unroll
    for (int mask = 1; mask < 16; mask <<= 1)
#pragma unroll
      for (int j = 0; j < 8; ++j) part[j] += __shfl_xor(part[j], mask, 32);
    if (var == 0) {
#pragma unroll
      for (int j = 0; j < 8; ++j) res[j] = part[j];
    } else {
#pragma unroll
      for (int j = 0; j < 8; ++j) res[j] -= part[j];  // b2ot cancels
    }
  }
  if (cn == 0) {
#pragma unroll
    for (int j = 0; j < 8; ++j)
      out[q0 + wv * 16 + j + 8 * ch] = res[j];
  }
}

// ---------------------------------------------------------------------------
// host side
// ---------------------------------------------------------------------------
extern "C" void kernel_launch(void* const* d_in, const int* in_sizes, int n_in,
                              void* d_out, int out_size, void* d_ws, size_t ws_size,
                              hipStream_t stream) {
  (void)in_sizes; (void)n_in; (void)out_size; (void)ws_size;
  const int N = 50000, E = 1600000, Q = 400000;

  const int* batch = (const int*)d_in[0];
  const float* x = (const float*)d_in[1];
  const int* ei = (const int*)d_in[2];
  const int* esrc = ei;
  const int* edst = ei + E;
  const int* srci = (const int*)d_in[3];
  const int* dsti = (const int*)d_in[4];
  auto F = [&](int i) { return (const float*)d_in[i]; };

  // d_in param indices assume recursive dict-insertion-order flattening:
  // 5..12 emb_mlp{lin_in.W,b, hid.W,b, lin_out.W,b, proj.W,b}
  // 13..20 emb_rho same; 21..29 conv{1,2,3}{lin_l.W,b, lin_r.W}
  // 30..37 lin1; 38..43 lin2{lin_in.W,b, hid.W,b, lin_out.W,b}

  _Float16* wbase = (_Float16*)d_ws;
  size_t oh = 0;
  auto A = [&](size_t nel) { size_t r = oh; oh += (nel + 63) & ~(size_t)63; return r; };
  size_t oWein = A(64 * 512), oWehd = A(64 * 64), oWeot = A(512 * 64), oWepr = A(64 * 512);
  size_t oWrin = A(64 * 512), oWrhd = A(64 * 64), oWrot = A(64 * 64), oWrpr = A(64 * 512);
  size_t oCl[3], oCr[3];
  for (int l = 0; l < 3; ++l) { oCl[l] = A(64 * 64); oCr[l] = A(64 * 64); }
  size_t oW1in = A(128 * 192), oW1hd = A(128 * 128), oW1ot = A(128 * 128), oW1pr = A(128 * 192);
  size_t oW2in = A(128 * 128), oW2hd = A(128 * 128);

  size_t fbytes = (oh * sizeof(_Float16) + 255) & ~(size_t)255;
  char* cb = (char*)d_ws + fbytes;
  float* hA = (float*)cb;       cb += (size_t)N * 64 * 4;
  float* hB = (float*)cb;       cb += (size_t)N * 64 * 4;
  float* agg = (float*)cb;      cb += (size_t)N * 64 * 4;
  float* deg = (float*)cb;      cb += (size_t)N * 4;
  unsigned* pu = (unsigned*)cb; cb += 64 * 64 * 4;
  float* gemb = (float*)cb;     cb += 64 * 64 * 4;

  auto cvt = [&](int idx, size_t off, int nel) {
    k_cvt<<<(nel + 255) / 256, 256, 0, stream>>>(F(idx), wbase + off, nel);
  };
  cvt(5, oWein, 64 * 512);  cvt(7, oWehd, 64 * 64);
  cvt(9, oWeot, 512 * 64);  cvt(11, oWepr, 64 * 512);
  cvt(13, oWrin, 64 * 512); cvt(15, oWrhd, 64 * 64);
  cvt(17, oWrot, 64 * 64);  cvt(19, oWrpr, 64 * 512);
  const int convW[3][2] = {{21, 23}, {24, 26}, {27, 29}};
  for (int l = 0; l < 3; ++l) {
    cvt(convW[l][0], oCl[l], 64 * 64);
    cvt(convW[l][1], oCr[l], 64 * 64);
  }
  cvt(30, oW1in, 128 * 192); cvt(32, oW1hd, 128 * 128);
  cvt(34, oW1ot, 128 * 128); cvt(36, oW1pr, 128 * 192);
  cvt(38, oW2in, 128 * 128); cvt(40, oW2hd, 128 * 128);

  // embedder (16 nodes per block)
  k_embed<<<N / 16, 128, 0, stream>>>(
      x,
      wbase + oWein, wbase + oWepr, wbase + oWehd, wbase + oWeot,
      F(6), F(12), F(8), F(10),
      wbase + oWrin, wbase + oWrpr, wbase + oWrhd, wbase + oWrot,
      F(14), F(20), F(16), F(18),
      hA);

  // degree
  k_zero<<<(N + 255) / 256, 256, 0, stream>>>(deg, N);
  k_deg<<<(E + 255) / 256, 256, 0, stream>>>(edst, deg, E);

  // 3 SAGE layers
  float* hin = hA;
  float* hout = hB;
  const int blIdx[3] = {22, 25, 28};
  for (int l = 0; l < 3; ++l) {
    k_zero<<<((N * 64) + 255) / 256, 256, 0, stream>>>(agg, N * 64);
    k_scatter<<<((E * 8) + 255) / 256, 256, 0, stream>>>(esrc, edst, hin, agg, E);
    k_conv<<<(N + 63) / 64, 128, 0, stream>>>(hin, agg, deg,
                                              wbase + oCl[l], F(blIdx[l]),
                                              wbase + oCr[l], hout, N);
    float* t = hin; hin = hout; hout = t;
  }

  // graph pooling (segment_max)
  k_pool_init<<<16, 256, 0, stream>>>(pu);
  k_pool<<<((N * 64) + 255) / 256, 256, 0, stream>>>(hin, batch, pu, N);
  k_pool_dec<<<16, 256, 0, stream>>>(pu, gemb);

  // head
  k_head<<<Q / 64, 128, 0, stream>>>(
      hin, gemb, srci, dsti, batch,
      wbase + oW1in, F(31), wbase + oW1pr, F(37),
      wbase + oW1hd, F(33), wbase + oW1ot, F(35),
      wbase + oW2in, F(39), wbase + oW2hd, F(41),
      F(42),
      (float*)d_out);
}
